// FlowNet3DLegacy_80556406604132
// MI455X (gfx1250) — matrix operand order repaired and checked by reading, not compiled, not run on hardware
//
#include <hip/hip_runtime.h>
#include <hip/hip_bf16.h>
#include <stdint.h>

// FlowNet3D forward on gfx1250 (MI455X).
// All 1x1-conv GEMMs run on v_wmma_f32_16x16x32_bf16 with f32 accumulation.
// Activations/weights are pre-converted to zero-padded bf16 buffers so the
// WMMA K-loop is pure b128 loads + WMMA (no guards, no conversion ALU).
// Requires ws_size >= ~340 MB.

#define BATCH 8
#define NPTS  4096
#define TPB   256

static inline int cdiv(int a, int b) { return (a + b - 1) / b; }
static inline int align32(int a) { return (a + 31) & ~31; }

typedef __attribute__((ext_vector_type(16))) __bf16 v16bf;
typedef __attribute__((ext_vector_type(8)))  __bf16 v8bf;
typedef __attribute__((ext_vector_type(8)))  float  v8f;

__device__ __forceinline__ __bf16 f2bf(float f) {
  uint32_t u = __builtin_bit_cast(uint32_t, f);
  uint32_t r = u + 0x7FFFu + ((u >> 16) & 1u);   // round-to-nearest-even
  uint16_t h = (uint16_t)(r >> 16);
  return __builtin_bit_cast(__bf16, h);
}

// ---------------------------------------------------------------- transpose [B,3,N] -> [B,N,3]
__global__ void k_transpose_c3(const float* __restrict__ in, float* __restrict__ out,
                               int Bn, int Np) {
  int t = blockIdx.x * blockDim.x + threadIdx.x;
  if (t >= Bn * Np) return;
  int b = t / Np, n = t % Np;
  const float* src = in + (size_t)b * 3 * Np;
  float* dst = out + (size_t)t * 3;
  dst[0] = src[0 * Np + n];
  dst[1] = src[1 * Np + n];
  dst[2] = src[2 * Np + n];
}

// ---------------------------------------------------------------- farthest point sampling
// one block per batch; dist[] lives in LDS (n <= 4096 -> 16KB of 320KB WGP pool)
__global__ void k_fps(const float* __restrict__ xyz, int* __restrict__ out,
                      int n, int npoint) {
  int b = blockIdx.x;
  const float* X = xyz + (size_t)b * n * 3;
  extern __shared__ float smem[];
  float* dist = smem;                       // n floats
  float* rmax = smem + n;                   // blockDim floats
  int*   rarg = (int*)(rmax + blockDim.x);  // blockDim ints
  for (int i = threadIdx.x; i < n; i += blockDim.x) dist[i] = 1e10f;
  __syncthreads();
  int far = 0;
  for (int s = 0; s < npoint; ++s) {
    if (threadIdx.x == 0) out[b * npoint + s] = far;
    float cx = X[far * 3 + 0], cy = X[far * 3 + 1], cz = X[far * 3 + 2];
    float best = -1.0f; int barg = 0;
    for (int i = threadIdx.x; i < n; i += blockDim.x) {
      float dx = X[i * 3 + 0] - cx, dy = X[i * 3 + 1] - cy, dz = X[i * 3 + 2] - cz;
      float d = dx * dx + dy * dy + dz * dz;
      float m = fminf(dist[i], d);
      dist[i] = m;
      if (m > best) { best = m; barg = i; }
    }
    rmax[threadIdx.x] = best; rarg[threadIdx.x] = barg;
    __syncthreads();
    for (int off = blockDim.x >> 1; off > 0; off >>= 1) {
      if (threadIdx.x < off) {
        float ov = rmax[threadIdx.x + off]; int oi = rarg[threadIdx.x + off];
        if (ov > rmax[threadIdx.x] ||
            (ov == rmax[threadIdx.x] && oi < rarg[threadIdx.x])) {
          rmax[threadIdx.x] = ov; rarg[threadIdx.x] = oi;
        }
      }
      __syncthreads();
    }
    far = rarg[0];
    __syncthreads();
  }
}

// ---------------------------------------------------------------- gather xyz by index
__global__ void k_gather3(const float* __restrict__ pts, const int* __restrict__ idx,
                          float* __restrict__ out, int Bn, int n, int S) {
  int t = blockIdx.x * blockDim.x + threadIdx.x;
  if (t >= Bn * S) return;
  int b = t / S;
  int i = idx[t];
  const float* src = pts + ((size_t)b * n + i) * 3;
  float* dst = out + (size_t)t * 3;
  dst[0] = src[0]; dst[1] = src[1]; dst[2] = src[2];
}

// ---------------------------------------------------------------- ball query (first K in-ball, backfill first)
__global__ void k_query_ball(const float* __restrict__ xyz, const float* __restrict__ nxyz,
                             int* __restrict__ out, int Bn, int n, int S, int K, float r2) {
  int t = blockIdx.x * blockDim.x + threadIdx.x;
  if (t >= Bn * S) return;
  int b = t / S;
  const float* c  = nxyz + (size_t)t * 3;
  const float* pb = xyz + (size_t)b * n * 3;
  float cx = c[0], cy = c[1], cz = c[2];
  int cnt = 0, first = -1;
  for (int i = 0; i < n && cnt < K; ++i) {
    float dx = pb[i * 3 + 0] - cx, dy = pb[i * 3 + 1] - cy, dz = pb[i * 3 + 2] - cz;
    float d = dx * dx + dy * dy + dz * dz;
    if (d <= r2) {
      if (first < 0) first = i;
      out[(size_t)t * K + cnt] = i;
      ++cnt;
    }
  }
  if (first < 0) first = 0;
  for (int j = cnt; j < K; ++j) out[(size_t)t * K + j] = first;
}

// ---------------------------------------------------------------- kNN (k <= 64), insertion sort per query
__global__ void k_knn(const float* __restrict__ q, const float* __restrict__ p,
                      int* __restrict__ oidx, float* __restrict__ odist,
                      int Bn, int Q, int P, int k) {
  int t = blockIdx.x * blockDim.x + threadIdx.x;
  if (t >= Bn * Q) return;
  int b = t / Q;
  const float* qp = q + (size_t)t * 3;
  float qx = qp[0], qy = qp[1], qz = qp[2];
  float bd[64]; int bi[64];
  for (int j = 0; j < k; ++j) { bd[j] = 3.0e38f; bi[j] = 0; }
  const float* pb = p + (size_t)b * P * 3;
  for (int i = 0; i < P; ++i) {
    float dx = pb[i * 3 + 0] - qx, dy = pb[i * 3 + 1] - qy, dz = pb[i * 3 + 2] - qz;
    float d = dx * dx + dy * dy + dz * dz;
    if (d < bd[k - 1]) {
      int j = k - 1;
      while (j > 0 && bd[j - 1] > d) { bd[j] = bd[j - 1]; bi[j] = bi[j - 1]; --j; }
      bd[j] = d; bi[j] = i;
    }
  }
  for (int j = 0; j < k; ++j) {
    oidx[(size_t)t * k + j] = bi[j];
    if (odist) odist[(size_t)t * k + j] = bd[j];
  }
}

// ---------------------------------------------------------------- grouping kernels
__global__ void k_group_sa(const float* __restrict__ xyz, const float* __restrict__ nxyz,
                           const float* __restrict__ feats, const int* __restrict__ gi,
                           float* __restrict__ X, int Bn, int n, int S, int K, int Cf) {
  int t = blockIdx.x * blockDim.x + threadIdx.x;
  if (t >= Bn * S * K) return;
  int s = (t / K) % S;
  int b = t / (K * S);
  int g = gi[t];
  const float* pp = xyz + ((size_t)b * n + g) * 3;
  const float* cc = nxyz + ((size_t)b * S + s) * 3;
  float* xr = X + (size_t)t * (3 + Cf);
  xr[0] = pp[0] - cc[0]; xr[1] = pp[1] - cc[1]; xr[2] = pp[2] - cc[2];
  const float* fr = feats + ((size_t)b * n + g) * Cf;
  for (int c = 0; c < Cf; ++c) xr[3 + c] = fr[c];
}

__global__ void k_group_fe(const float* __restrict__ pos1, const float* __restrict__ pos2,
                           const float* __restrict__ f1, const float* __restrict__ f2,
                           const int* __restrict__ idx, float* __restrict__ X,
                           int Bn, int S, int P, int K, int C) {
  int t = blockIdx.x * blockDim.x + threadIdx.x;
  if (t >= Bn * S * K) return;
  int s = (t / K) % S;
  int b = t / (K * S);
  int i = idx[t];
  const float* pq = pos1 + ((size_t)b * S + s) * 3;
  const float* pp = pos2 + ((size_t)b * P + i) * 3;
  float* xr = X + (size_t)t * (3 + 2 * C);
  xr[0] = pp[0] - pq[0]; xr[1] = pp[1] - pq[1]; xr[2] = pp[2] - pq[2];
  const float* g2 = f2 + ((size_t)b * P + i) * C;
  const float* g1 = f1 + ((size_t)b * S + s) * C;
  for (int c = 0; c < C; ++c) xr[3 + c] = g2[c];
  for (int c = 0; c < C; ++c) xr[3 + C + c] = g1[c];
}

__global__ void k_group_su(const float* __restrict__ pos1, const float* __restrict__ pos2,
                           const float* __restrict__ f2, const int* __restrict__ idx,
                           float* __restrict__ X, int Bn, int S, int P, int K, int C) {
  int t = blockIdx.x * blockDim.x + threadIdx.x;
  if (t >= Bn * S * K) return;
  int s = (t / K) % S;
  int b = t / (K * S);
  int i = idx[t];
  const float* pq = pos1 + ((size_t)b * S + s) * 3;
  const float* pp = pos2 + ((size_t)b * P + i) * 3;
  float* xr = X + (size_t)t * (C + 3);
  const float* g2 = f2 + ((size_t)b * P + i) * C;
  for (int c = 0; c < C; ++c) xr[c] = g2[c];
  xr[C + 0] = pp[0] - pq[0]; xr[C + 1] = pp[1] - pq[1]; xr[C + 2] = pp[2] - pq[2];
}

// ---------------------------------------------------------------- bf16 conversion (padded)
// f32 [M,K] -> bf16 [M,Kp], zero pad (Kp % 32 == 0)
__global__ void k_cvt_pad(const float* __restrict__ X, __bf16* __restrict__ Xb,
                          int M, int K, int Kp) {
  int t = blockIdx.x * blockDim.x + threadIdx.x;
  if (t >= M * Kp) return;
  int r = t / Kp, c = t % Kp;
  Xb[t] = (c < K) ? f2bf(X[(size_t)r * K + c]) : f2bf(0.0f);
}

// W f32 [K,N] -> Wt bf16 [N,Kp] (transposed, zero pad)
__global__ void k_cvt_w(const float* __restrict__ W, __bf16* __restrict__ Wt,
                        int K, int N, int Kp) {
  int t = blockIdx.x * blockDim.x + threadIdx.x;
  if (t >= N * Kp) return;
  int n = t / Kp, k = t % Kp;
  Wt[t] = (k < K) ? f2bf(W[(size_t)k * N + n]) : f2bf(0.0f);
}

// ---------------------------------------------------------------- GEMM: Y[M,N] = A[M,Kp] * Wt[N,Kp]^T
// bf16 WMMA 16x16x32, f32 accumulate. One wave computes a 16x32 output tile
// (two 16x16 N-tiles reusing the A fragment); 8 waves/block cover 128 rows.
// Preconditions: M % 128 == 0, N % 32 == 0, Kp % 32 == 0 (padded buffers).
// Fragment layouts used here (A/B in contiguous k order, not the ISA table order):
//   A 16x32:  lane half hl, elements 0..7  <- K = 16*hl + 0..7
//                           elements 8..15 <- K = 16*hl + 8..15   (two b128 loads)
//   B 32x16:  elements 0..15 <- K = 16*hl + 0..15 at fixed N = lane%16 (contiguous in Wt row)
//   D 16x16:  VGPR v -> M = v + 8*hl, N = lane%16
__global__ __launch_bounds__(256)
void k_gemm_bf16(const __bf16* __restrict__ A, const __bf16* __restrict__ Wt,
                 float* __restrict__ Y, int M, int Kp, int N) {
  int wave = threadIdx.x >> 5;
  int lane = threadIdx.x & 31;
  int tM = blockIdx.x * 8 + wave;
  int lid = lane & 15;
  int hl  = lane >> 4;
  int rowA = tM * 16 + lid;
  int col0 = blockIdx.y * 32 + lid;
  const __bf16* Ap  = A  + (size_t)rowA * Kp + 16 * hl;
  const __bf16* Bp0 = Wt + (size_t)col0 * Kp + 16 * hl;
  const __bf16* Bp1 = Wt + (size_t)(col0 + 16) * Kp + 16 * hl;
  v8f acc0 = {};
  v8f acc1 = {};
  for (int k0 = 0; k0 < Kp; k0 += 32) {
    v8bf alo = *(const v8bf*)(Ap + k0);
    v8bf ahi = *(const v8bf*)(Ap + k0 + 8);
    v16bf av = __builtin_shufflevector(alo, ahi,
                                       0, 1, 2, 3, 4, 5, 6, 7,
                                       8, 9, 10, 11, 12, 13, 14, 15);
    v16bf b0 = *(const v16bf*)(Bp0 + k0);
    v16bf b1 = *(const v16bf*)(Bp1 + k0);
    acc0 = __builtin_amdgcn_wmma_f32_16x16x32_bf16(false, av, false, b0,
                                                   (short)0, acc0, false, false);
    acc1 = __builtin_amdgcn_wmma_f32_16x16x32_bf16(false, av, false, b1,
                                                   (short)0, acc1, false, false);
  }
#pragma unroll
  for (int v = 0; v < 8; ++v) {
    int r = tM * 16 + v + 8 * hl;
    Y[(size_t)r * N + col0]      = acc0[v];
    Y[(size_t)r * N + col0 + 16] = acc1[v];
  }
}

// ---------------------------------------------------------------- batchnorm (train-mode stats) + relu
__global__ void k_bn_stats(const float* __restrict__ Y, int M, int N,
                           float* __restrict__ mu, float* __restrict__ var) {
  int c = blockIdx.x;
  float s = 0.0f, s2 = 0.0f;
  for (int r = threadIdx.x; r < M; r += blockDim.x) {
    float v = Y[(size_t)r * N + c];
    s += v; s2 += v * v;
  }
  __shared__ float sh[TPB], sh2[TPB];
  sh[threadIdx.x] = s; sh2[threadIdx.x] = s2;
  __syncthreads();
  for (int o = blockDim.x >> 1; o > 0; o >>= 1) {
    if (threadIdx.x < o) { sh[threadIdx.x] += sh[threadIdx.x + o]; sh2[threadIdx.x] += sh2[threadIdx.x + o]; }
    __syncthreads();
  }
  if (threadIdx.x == 0) {
    float m = sh[0] / (float)M;
    mu[c] = m;
    var[c] = sh2[0] / (float)M - m * m;
  }
}

// normalize + relu in place (f32) and emit bf16 copy for the next GEMM (stride N, N%32==0)
__global__ void k_bn_relu_cvt(float* __restrict__ Y, __bf16* __restrict__ Yb,
                              int total, int N,
                              const float* __restrict__ mu, const float* __restrict__ var,
                              const float* __restrict__ g, const float* __restrict__ b) {
  int t = blockIdx.x * blockDim.x + threadIdx.x;
  if (t >= total) return;
  int c = t % N;
  float y = (Y[t] - mu[c]) * rsqrtf(var[c] + 1e-5f) * g[c] + b[c];
  y = fmaxf(y, 0.0f);
  Y[t] = y;
  Yb[t] = f2bf(y);
}

// ---------------------------------------------------------------- max pool over K neighbors
__global__ void k_maxpool(const float* __restrict__ X, float* __restrict__ out,
                          int R, int K, int C) {
  int t = blockIdx.x * blockDim.x + threadIdx.x;
  if (t >= R * C) return;
  int r = t / C, c = t % C;
  float m = -3.0e38f;
  for (int k = 0; k < K; ++k) m = fmaxf(m, X[((size_t)r * K + k) * C + c]);
  out[(size_t)r * C + c] = m;
}

// ---------------------------------------------------------------- channel concat out[R,C1+C2] = [A|B]
__global__ void k_concat2(const float* __restrict__ A, const float* __restrict__ Bm,
                          float* __restrict__ out, int R, int C1, int C2) {
  int t = blockIdx.x * blockDim.x + threadIdx.x;
  int C = C1 + C2;
  if (t >= R * C) return;
  int r = t / C, c = t % C;
  out[t] = (c < C1) ? A[(size_t)r * C1 + c] : Bm[(size_t)r * C2 + (c - C1)];
}

// ---------------------------------------------------------------- feature-prop interpolation -> [interp(C2), f1(3)]
__global__ void k_fp_interp(const float* __restrict__ f2, const float* __restrict__ f1,
                            const int* __restrict__ idx, const float* __restrict__ dist,
                            float* __restrict__ X, int Bn, int Np, int P, int C2) {
  int t = blockIdx.x * blockDim.x + threadIdx.x;
  if (t >= Bn * Np) return;
  int b = t / Np;
  const int* id = idx + (size_t)t * 3;
  const float* dd = dist + (size_t)t * 3;
  float w0 = 1.0f / fmaxf(dd[0], 1e-10f);
  float w1 = 1.0f / fmaxf(dd[1], 1e-10f);
  float w2 = 1.0f / fmaxf(dd[2], 1e-10f);
  float s = w0 + w1 + w2; w0 /= s; w1 /= s; w2 /= s;
  const float* a0 = f2 + ((size_t)b * P + id[0]) * C2;
  const float* a1 = f2 + ((size_t)b * P + id[1]) * C2;
  const float* a2 = f2 + ((size_t)b * P + id[2]) * C2;
  float* xr = X + (size_t)t * (C2 + 3);
  for (int c = 0; c < C2; ++c) xr[c] = w0 * a0[c] + w1 * a1[c] + w2 * a2[c];
  const float* fr = f1 + (size_t)t * 3;
  xr[C2 + 0] = fr[0]; xr[C2 + 1] = fr[1]; xr[C2 + 2] = fr[2];
}

// ---------------------------------------------------------------- head2: [B*N,128] x [128,3] + b -> d_out [B,3,N]
__global__ void k_head2(const float* __restrict__ X, const float* __restrict__ W,
                        const float* __restrict__ bias, float* __restrict__ out,
                        int Bn, int Np) {
  int t = blockIdx.x * blockDim.x + threadIdx.x;
  if (t >= Bn * Np) return;
  int b = t / Np, n = t % Np;
  const float* xr = X + (size_t)t * 128;
  for (int d = 0; d < 3; ++d) {
    float acc = bias[d];
    for (int c = 0; c < 128; ++c) acc += xr[c] * W[c * 3 + d];
    out[((size_t)b * 3 + d) * Np + n] = acc;
  }
}

// ================================================================ host orchestration
struct Layer { const float* W; const float* g; const float* b; int cin, cout; };

// Run an MLP over Xf [M,Cin] (f32). Intermediate f32 layers overwrite Xf; the
// final layer's f32 output goes to finalY. hA/hB are padded bf16 ping-pong
// activation buffers; wt is the bf16 transposed-weight scratch.
static void run_mlp(float* Xf, int M, int Cin, const Layer* mlp, int nl,
                    float* finalY, __bf16* hA, __bf16* hB, __bf16* wt,
                    float* mu, float* var, hipStream_t st) {
  if (nl == 0) return;
  int Kp = align32(Cin);
  k_cvt_pad<<<cdiv(M * Kp, TPB), TPB, 0, st>>>(Xf, hA, M, Cin, Kp);
  const __bf16* src = hA;
  __bf16* dsth = hB;
  for (int i = 0; i < nl; ++i) {
    const Layer& L = mlp[i];
    int KpW = align32(L.cin);
    k_cvt_w<<<cdiv(L.cout * KpW, TPB), TPB, 0, st>>>(L.W, wt, L.cin, L.cout, KpW);
    float* Ydst = (i == nl - 1) ? finalY : Xf;
    dim3 grid(cdiv(M, 128), cdiv(L.cout, 32));
    k_gemm_bf16<<<grid, 256, 0, st>>>(src, wt, Ydst, M, KpW, L.cout);
    k_bn_stats<<<L.cout, TPB, 0, st>>>(Ydst, M, L.cout, mu, var);
    int tot = M * L.cout;
    k_bn_relu_cvt<<<cdiv(tot, TPB), TPB, 0, st>>>(Ydst, dsth, tot, L.cout, mu, var, L.g, L.b);
    const __bf16* t2 = src; src = dsth; dsth = (__bf16*)t2;
  }
}

static void set_abstraction(const float* xyz, int n, const float* feats, int Cf,
                            int S, float radius, int K, const Layer* mlp, int nl,
                            float* new_xyz, float* out_feats,
                            int* fps_idx, int* gi, float* bufA,
                            __bf16* hA, __bf16* hB, __bf16* wt,
                            float* mu, float* var, hipStream_t st) {
  size_t shmem = (size_t)n * 4 + (size_t)TPB * 8;
  k_fps<<<BATCH, TPB, shmem, st>>>(xyz, fps_idx, n, S);
  k_gather3<<<cdiv(BATCH * S, TPB), TPB, 0, st>>>(xyz, fps_idx, new_xyz, BATCH, n, S);
  k_query_ball<<<cdiv(BATCH * S, TPB), TPB, 0, st>>>(xyz, new_xyz, gi, BATCH, n, S, K,
                                                     radius * radius);
  int rows = BATCH * S * K;
  k_group_sa<<<cdiv(rows, TPB), TPB, 0, st>>>(xyz, new_xyz, feats, gi, bufA,
                                              BATCH, n, S, K, Cf);
  run_mlp(bufA, rows, 3 + Cf, mlp, nl, bufA, hA, hB, wt, mu, var, st);
  int C = mlp[nl - 1].cout;
  k_maxpool<<<cdiv(BATCH * S * C, TPB), TPB, 0, st>>>(bufA, out_feats, BATCH * S, K, C);
}

static void set_upconv(const float* pos1, int S, const float* pos2, int P,
                       const float* f1, int C1, const float* f2, int C2, int K,
                       const Layer* m1, int n1, const Layer* m2, int n2,
                       float* out, int* kidx, float* pooled, float* bufA,
                       __bf16* hA, __bf16* hB, __bf16* wt,
                       float* mu, float* var, hipStream_t st) {
  k_knn<<<cdiv(BATCH * S, TPB), TPB, 0, st>>>(pos1, pos2, kidx, (float*)nullptr,
                                              BATCH, S, P, K);
  int rows = BATCH * S * K;
  k_group_su<<<cdiv(rows, TPB), TPB, 0, st>>>(pos1, pos2, f2, kidx, bufA,
                                              BATCH, S, P, K, C2);
  run_mlp(bufA, rows, C2 + 3, m1, n1, bufA, hA, hB, wt, mu, var, st);
  int Cp = n1 ? m1[n1 - 1].cout : C2 + 3;
  k_maxpool<<<cdiv(BATCH * S * Cp, TPB), TPB, 0, st>>>(bufA, pooled, BATCH * S, K, Cp);
  int Cc = Cp + C1;
  k_concat2<<<cdiv(BATCH * S * Cc, TPB), TPB, 0, st>>>(pooled, f1, bufA, BATCH * S, Cp, C1);
  run_mlp(bufA, BATCH * S, Cc, m2, n2, out, hA, hB, wt, mu, var, st);
}

extern "C" void kernel_launch(void* const* d_in, const int* in_sizes, int n_in,
                              void* d_out, int out_size, void* d_ws, size_t ws_size,
                              hipStream_t stream) {
  (void)in_sizes; (void)n_in; (void)out_size; (void)ws_size;
  const float* pc1 = (const float*)d_in[0];
  const float* pc2 = (const float*)d_in[1];
  const float* fe1 = (const float*)d_in[2];
  const float* fe2 = (const float*)d_in[3];
  int pi = 4;
  auto layer = [&](int cin, int cout) {
    Layer L;
    L.W = (const float*)d_in[pi++];
    L.g = (const float*)d_in[pi++];
    L.b = (const float*)d_in[pi++];
    L.cin = cin; L.cout = cout;
    return L;
  };
  Layer sa1[3]   = { layer(6, 32),    layer(32, 32),   layer(32, 64)   };
  Layer sa2[3]   = { layer(67, 64),   layer(64, 64),   layer(64, 128)  };
  Layer sa3[3]   = { layer(259, 128), layer(128, 128), layer(128, 256) };
  Layer sa4[3]   = { layer(259, 256), layer(256, 256), layer(256, 512) };
  Layer feL[1]   = { layer(259, 256) };
  // su1_mlp1 is empty
  Layer su1m2[2] = { layer(771, 256), layer(256, 256) };
  Layer su2m1[3] = { layer(259, 128), layer(128, 128), layer(128, 256) };
  Layer su2m2[1] = { layer(640, 256) };
  Layer su3m1[3] = { layer(259, 128), layer(128, 128), layer(128, 256) };
  Layer su3m2[1] = { layer(320, 256) };
  Layer fpL[2]   = { layer(259, 256), layer(256, 256) };
  Layer head1[1] = { layer(256, 128) };
  const float* h2W = (const float*)d_in[pi++];
  const float* h2b = (const float*)d_in[pi++];

  // ---- workspace bump allocator (deterministic)
  float* ws = (float*)d_ws;
  size_t off = 0;
  auto alloc = [&](size_t nfl) -> float* {
    float* p = ws + off;
    off += (nfl + 63) & ~(size_t)63;
    return p;
  };
  float* p1   = alloc((size_t)BATCH * NPTS * 3);
  float* p2   = alloc((size_t)BATCH * NPTS * 3);
  float* f1c  = alloc((size_t)BATCH * NPTS * 3);
  float* f2c  = alloc((size_t)BATCH * NPTS * 3);
  float* l1p1 = alloc((size_t)BATCH * 1024 * 3);
  float* l1f1 = alloc((size_t)BATCH * 1024 * 64);
  float* l1p2 = alloc((size_t)BATCH * 1024 * 3);
  float* l1f2 = alloc((size_t)BATCH * 1024 * 64);
  float* l2p1 = alloc((size_t)BATCH * 256 * 3);
  float* l2f1 = alloc((size_t)BATCH * 256 * 128);
  float* l2p2 = alloc((size_t)BATCH * 256 * 3);
  float* l2f2 = alloc((size_t)BATCH * 256 * 128);
  float* l2f1n = alloc((size_t)BATCH * 256 * 256);
  float* l3p1 = alloc((size_t)BATCH * 64 * 3);
  float* l3f1 = alloc((size_t)BATCH * 64 * 256);
  float* l4p1 = alloc((size_t)BATCH * 16 * 3);
  float* l4f1 = alloc((size_t)BATCH * 16 * 512);
  float* l3fn = alloc((size_t)BATCH * 64 * 256);
  float* l2fn = alloc((size_t)BATCH * 256 * 256);
  float* l1fn = alloc((size_t)BATCH * 1024 * 256);
  float* catf1 = alloc((size_t)BATCH * 256 * 384);
  float* pooled = alloc((size_t)BATCH * 1024 * 256);
  float* mu  = alloc(1024);
  float* var = alloc(1024);
  float* fpdist = alloc((size_t)BATCH * NPTS * 3);
  int* fps_idx = (int*)alloc((size_t)BATCH * 1024);
  int* gi      = (int*)alloc((size_t)BATCH * 1024 * 16);
  int* kidx    = (int*)alloc((size_t)BATCH * 256 * 64);   // also covers fp (B*4096*3) & su3 (B*1024*8)
  float* bufA  = alloc(34000000);                          // f32 X/Y scratch; >= 131072*259
  __bf16* hA   = (__bf16*)alloc(19000000);                 // bf16 act ping; >= 131072*288/2
  __bf16* hB   = (__bf16*)alloc(19000000);                 // bf16 act pong
  __bf16* wt   = (__bf16*)alloc(120000);                   // bf16 W^T; >= 256*800/2

  // ---- transpose inputs to channels-last
  int tot = BATCH * NPTS;
  k_transpose_c3<<<cdiv(tot, TPB), TPB, 0, stream>>>(pc1, p1, BATCH, NPTS);
  k_transpose_c3<<<cdiv(tot, TPB), TPB, 0, stream>>>(pc2, p2, BATCH, NPTS);
  k_transpose_c3<<<cdiv(tot, TPB), TPB, 0, stream>>>(fe1, f1c, BATCH, NPTS);
  k_transpose_c3<<<cdiv(tot, TPB), TPB, 0, stream>>>(fe2, f2c, BATCH, NPTS);

  // ---- set abstraction levels 1,2 for both clouds
  set_abstraction(p1, NPTS, f1c, 3, 1024, 5.0f, 16, sa1, 3, l1p1, l1f1,
                  fps_idx, gi, bufA, hA, hB, wt, mu, var, stream);
  set_abstraction(l1p1, 1024, l1f1, 64, 256, 10.0f, 16, sa2, 3, l2p1, l2f1,
                  fps_idx, gi, bufA, hA, hB, wt, mu, var, stream);
  set_abstraction(p2, NPTS, f2c, 3, 1024, 5.0f, 16, sa1, 3, l1p2, l1f2,
                  fps_idx, gi, bufA, hA, hB, wt, mu, var, stream);
  set_abstraction(l1p2, 1024, l1f2, 64, 256, 10.0f, 16, sa2, 3, l2p2, l2f2,
                  fps_idx, gi, bufA, hA, hB, wt, mu, var, stream);

  // ---- flow embedding (knn=64, concat, mlp 259->256, maxpool)
  k_knn<<<cdiv(BATCH * 256, TPB), TPB, 0, stream>>>(l2p1, l2p2, kidx, (float*)nullptr,
                                                    BATCH, 256, 256, 64);
  int ferows = BATCH * 256 * 64;
  k_group_fe<<<cdiv(ferows, TPB), TPB, 0, stream>>>(l2p1, l2p2, l2f1, l2f2, kidx, bufA,
                                                    BATCH, 256, 256, 64, 128);
  run_mlp(bufA, ferows, 259, feL, 1, bufA, hA, hB, wt, mu, var, stream);
  k_maxpool<<<cdiv(BATCH * 256 * 256, TPB), TPB, 0, stream>>>(bufA, l2f1n, BATCH * 256, 64, 256);

  // ---- set abstraction levels 3,4 on cloud 1
  set_abstraction(l2p1, 256, l2f1n, 256, 64, 17.5f, 8, sa3, 3, l3p1, l3f1,
                  fps_idx, gi, bufA, hA, hB, wt, mu, var, stream);
  set_abstraction(l3p1, 64, l3f1, 256, 16, 25.0f, 8, sa4, 3, l4p1, l4f1,
                  fps_idx, gi, bufA, hA, hB, wt, mu, var, stream);

  // ---- set upconv stack
  set_upconv(l3p1, 64, l4p1, 16, l3f1, 256, l4f1, 512, 8,
             nullptr, 0, su1m2, 2, l3fn, kidx, pooled, bufA,
             hA, hB, wt, mu, var, stream);
  k_concat2<<<cdiv(BATCH * 256 * 384, TPB), TPB, 0, stream>>>(l2f1, l2f1n, catf1,
                                                              BATCH * 256, 128, 256);
  set_upconv(l2p1, 256, l3p1, 64, catf1, 384, l3fn, 256, 8,
             su2m1, 3, su2m2, 1, l2fn, kidx, pooled, bufA,
             hA, hB, wt, mu, var, stream);
  set_upconv(l1p1, 1024, l2p1, 256, l1f1, 64, l2fn, 256, 8,
             su3m1, 3, su3m2, 1, l1fn, kidx, pooled, bufA,
             hA, hB, wt, mu, var, stream);

  // ---- feature propagation to full resolution
  k_knn<<<cdiv(BATCH * NPTS, TPB), TPB, 0, stream>>>(p1, l1p1, kidx, fpdist,
                                                     BATCH, NPTS, 1024, 3);
  k_fp_interp<<<cdiv(BATCH * NPTS, TPB), TPB, 0, stream>>>(l1fn, f1c, kidx, fpdist, bufA,
                                                           BATCH, NPTS, 1024, 256);
  run_mlp(bufA, BATCH * NPTS, 259, fpL, 2, bufA, hA, hB, wt, mu, var, stream);

  // ---- head
  run_mlp(bufA, BATCH * NPTS, 256, head1, 1, bufA, hA, hB, wt, mu, var, stream);
  k_head2<<<cdiv(BATCH * NPTS, TPB), TPB, 0, stream>>>(bufA, h2W, h2b, (float*)d_out,
                                                       BATCH, NPTS);
}
